// GptOssSparseMoeBlock_17824114279000
// MI455X (gfx1250) — matrix operand order repaired and checked by reading, not compiled, not run on hardware
//
#include <hip/hip_runtime.h>

// ---------------------------------------------------------------------------
// GPT-OSS sparse MoE block for MI455X (gfx1250, wave32, WMMA bf16).
// T=512 tokens, H=2880, I=2880, E=8 experts, top-2 routing.
//
// HBM-bound on one-pass fp32 weight streaming (~800MB -> ~34us @ 23.3TB/s).
// Strategy: gather routed tokens per expert (4x FLOP reduction vs dense),
// convert fp32 operands to bf16 with v_cvt_pk_bf16_f32, run
// v_wmma_f32_16x16x32_bf16 from double-buffered LDS tiles (1 barrier / K-step),
// async global->LDS copies (ASYNCcnt) for the bf16 act operand, fused
// clamp/SiLU epilogue, weighted scatter-add via global f32 atomics.
//
// Workspace usage (< 6 MB):
//   [0)      topk_idx   int[1024]
//   [4096)   topk_w     float[1024]
//   [8192)   offs       int[16]
//   [12288)  slot_token int[1024]
//   [16384)  slot_w     float[1024]
//   [32768)  act        bf16[1024*2880]
// ---------------------------------------------------------------------------

#define T_TOK 512
#define H_DIM 2880
#define I_DIM 2880
#define E_NUM 8
#define ALPHA 1.702f
#define LIMIT 7.0f

typedef __bf16 bf16_t;
typedef bf16_t v16bf __attribute__((ext_vector_type(16)));
typedef float  v8f   __attribute__((ext_vector_type(8)));

union FragU {
    uint4 q[2];
    v16bf v;
};

#if defined(__has_builtin)
#if __has_builtin(__builtin_amdgcn_cvt_pk_bf16_f32)
#define USE_CVT_PK 1
#endif
#if __has_builtin(__builtin_amdgcn_global_load_async_to_lds_b128) && \
    __has_builtin(__builtin_amdgcn_s_wait_asynccnt)
#define USE_ASYNC_LDS 1
#endif
#endif

__device__ __forceinline__ unsigned bfround(float f) {
    unsigned u = __float_as_uint(f);
    return (u + 0x7FFFu + ((u >> 16) & 1u)) >> 16;   // RNE fp32 -> bf16 bits
}

__device__ __forceinline__ unsigned pk2(float lo, float hi) {
#ifdef USE_CVT_PK
    auto r = __builtin_amdgcn_cvt_pk_bf16_f32(lo, hi);  // v_cvt_pk_bf16_f32
    unsigned u;
    __builtin_memcpy(&u, &r, sizeof(u));
    return u;
#else
    return bfround(lo) | (bfround(hi) << 16);
#endif
}

#ifdef USE_ASYNC_LDS
// Builtin signature (from hipcc diagnostic): param0 is
// 'int __vector(4) __device__*' (AS1), so use GCC vector_size types with
// explicit numbered address spaces. Generic->AS pointers via integer
// round-trip: low 32 bits of a generic shared pointer are the LDS byte
// offset (ISA 10.2 aperture rules).
typedef int v4i_gcc __attribute__((vector_size(16)));
typedef v4i_gcc __attribute__((address_space(1)))* gbl_v4i_p;
typedef v4i_gcc __attribute__((address_space(3)))* lds_v4i_p;

__device__ __forceinline__ void async_cp16(const void* g, const void* l) {
    __builtin_amdgcn_global_load_async_to_lds_b128(
        (gbl_v4i_p)(uintptr_t)g,
        (lds_v4i_p)(unsigned int)(uintptr_t)l, 0, 0);
}
__device__ __forceinline__ void async_wait0() {
    __builtin_amdgcn_s_wait_asynccnt(0);
}
#endif

// ---------------------------------------------------------------------------
// Kernel 1: router logits + top-2 softmax. One wave32 per token.
// ---------------------------------------------------------------------------
__global__ void moe_router_kernel(const float* __restrict__ hs,
                                  const float* __restrict__ rw,
                                  const float* __restrict__ rb,
                                  int* __restrict__ topk_idx,
                                  float* __restrict__ topk_w) {
    const int wave = threadIdx.x >> 5;
    const int lane = threadIdx.x & 31;
    const int t = blockIdx.x * 8 + wave;
    if (t >= T_TOK) return;

    const float* hrow = hs + (size_t)t * H_DIM;
    float acc[E_NUM];
#pragma unroll
    for (int e = 0; e < E_NUM; ++e) acc[e] = 0.0f;

    for (int k = lane; k < H_DIM; k += 32) {
        float x = hrow[k];
#pragma unroll
        for (int e = 0; e < E_NUM; ++e)
            acc[e] = fmaf(x, rw[e * H_DIM + k], acc[e]);
    }
#pragma unroll
    for (int off = 16; off > 0; off >>= 1) {
#pragma unroll
        for (int e = 0; e < E_NUM; ++e)
            acc[e] += __shfl_xor(acc[e], off, 32);
    }
    if (lane == 0) {
        float lg[E_NUM];
#pragma unroll
        for (int e = 0; e < E_NUM; ++e) lg[e] = acc[e] + rb[e];
        int i0 = 0;
#pragma unroll
        for (int e = 1; e < E_NUM; ++e) if (lg[e] > lg[i0]) i0 = e;
        int i1 = (i0 == 0) ? 1 : 0;
#pragma unroll
        for (int e = 0; e < E_NUM; ++e)
            if (e != i0 && lg[e] > lg[i1]) i1 = e;
        float e1 = __expf(lg[i1] - lg[i0]);
        float inv = 1.0f / (1.0f + e1);
        topk_idx[2 * t + 0] = i0;
        topk_idx[2 * t + 1] = i1;
        topk_w[2 * t + 0] = inv;
        topk_w[2 * t + 1] = e1 * inv;
    }
}

// ---------------------------------------------------------------------------
// Kernel 2: deterministic per-expert token compaction (ballot prefix scan,
// token-ascending order, no atomics -> reproducible slot layout).
// ---------------------------------------------------------------------------
__global__ void moe_build_lists_kernel(const int* __restrict__ topk_idx,
                                       const float* __restrict__ topk_w,
                                       int* __restrict__ offs,
                                       int* __restrict__ slot_token,
                                       float* __restrict__ slot_w) {
    __shared__ int counts[E_NUM];
    __shared__ int offsS[E_NUM + 1];
    const int wave = threadIdx.x >> 5;
    const int lane = threadIdx.x & 31;
    const int e = wave;

    int cnt = 0;
    for (int base = 0; base < T_TOK; base += 32) {
        int t = base + lane;
        int i0 = topk_idx[2 * t + 0];
        int i1 = topk_idx[2 * t + 1];
        bool m = (i0 == e) || (i1 == e);
        unsigned long long bal = __ballot(m) & 0xFFFFFFFFull;
        cnt += __popcll(bal);
    }
    if (lane == 0) counts[e] = cnt;
    __syncthreads();
    if (threadIdx.x == 0) {
        int s = 0;
        for (int k = 0; k < E_NUM; ++k) { offsS[k] = s; s += counts[k]; }
        offsS[E_NUM] = s;
        for (int k = 0; k <= E_NUM; ++k) offs[k] = offsS[k];
    }
    __syncthreads();

    int pos = offsS[e];
    for (int base = 0; base < T_TOK; base += 32) {
        int t = base + lane;
        int i0 = topk_idx[2 * t + 0];
        int i1 = topk_idx[2 * t + 1];
        bool m = (i0 == e) || (i1 == e);
        unsigned long long bal = __ballot(m) & 0xFFFFFFFFull;
        int pre = __popcll(bal & ((1ull << lane) - 1ull));
        if (m) {
            int slot = pos + pre;
            slot_token[slot] = t;
            slot_w[slot] = (i0 == e) ? topk_w[2 * t + 0] : topk_w[2 * t + 1];
        }
        pos += __popcll(bal);
    }
}

// ---------------------------------------------------------------------------
// Kernel 3: gate_up GEMM + fused gated activation -> bf16 act buffer.
// 128 gathered tokens x (64 gate + matched 64 up cols), double-buffered LDS,
// one barrier per 32-wide K-step.
// ---------------------------------------------------------------------------
__global__ void __launch_bounds__(256)
moe_gemm_gateup_kernel(const float* __restrict__ hs,
                       const float* __restrict__ wgu,
                       const float* __restrict__ bgu,
                       const int* __restrict__ offs,
                       const int* __restrict__ slot_token,
                       unsigned short* __restrict__ act) {
    const int e  = blockIdx.z;
    const int mt = blockIdx.y;
    const int n0 = blockIdx.x * 64;

    const int base = offs[e];
    const int ce   = offs[e + 1] - base;
    if (mt * 128 >= ce) return;
    const int mbase = base + mt * 128;
    const int rows_valid = min(128, ce - mt * 128);

    __shared__ unsigned short As[2][128][40];   // 2 x 10 KB
    __shared__ unsigned short Bs[2][128][40];   // rows 0-63 gate, 64-127 up

    const int ltid  = threadIdx.x;
    const int lrow  = ltid >> 1;
    const int lhalf = ltid & 1;

    int tok = 0;
    if (lrow < rows_valid) tok = slot_token[mbase + lrow];
    const float* aSrc = hs + (size_t)tok * H_DIM + lhalf * 16;

    const int wrow = (lrow < 64) ? (n0 + lrow) : (I_DIM + n0 + (lrow - 64));
    const float* bSrc = wgu + ((size_t)e * 2 * I_DIM + wrow) * H_DIM + lhalf * 16;

    const int wv   = ltid >> 5;
    const int lane = ltid & 31;
    const int mq   = wv & 3;
    const int nq   = wv >> 2;
    const int lr   = lane & 15;
    const int hl   = lane >> 4;

    auto stage = [&](int k0, int b) {
        {
            const float4* s = reinterpret_cast<const float4*>(aSrc + k0);
            float4 x0 = s[0], x1 = s[1], x2 = s[2], x3 = s[3];
            uint4 p0 = make_uint4(pk2(x0.x, x0.y), pk2(x0.z, x0.w),
                                  pk2(x1.x, x1.y), pk2(x1.z, x1.w));
            uint4 p1 = make_uint4(pk2(x2.x, x2.y), pk2(x2.z, x2.w),
                                  pk2(x3.x, x3.y), pk2(x3.z, x3.w));
            uint4* d = reinterpret_cast<uint4*>(&As[b][lrow][lhalf * 16]);
            d[0] = p0; d[1] = p1;
        }
        {
            const float4* s = reinterpret_cast<const float4*>(bSrc + k0);
            float4 x0 = s[0], x1 = s[1], x2 = s[2], x3 = s[3];
            uint4 p0 = make_uint4(pk2(x0.x, x0.y), pk2(x0.z, x0.w),
                                  pk2(x1.x, x1.y), pk2(x1.z, x1.w));
            uint4 p1 = make_uint4(pk2(x2.x, x2.y), pk2(x2.z, x2.w),
                                  pk2(x3.x, x3.y), pk2(x3.z, x3.w));
            uint4* d = reinterpret_cast<uint4*>(&Bs[b][lrow][lhalf * 16]);
            d[0] = p0; d[1] = p1;
        }
    };

    v8f accg[2][2] = {};
    v8f accu[2][2] = {};

    stage(0, 0);
    int buf = 0;
    for (int k0 = 0; k0 < H_DIM; k0 += 32) {
        __syncthreads();                 // staged tile in 'buf' is visible
        const int nk = k0 + 32;
        if (nk < H_DIM) {
            __builtin_prefetch(bSrc + nk + 32, 0, 1);
            stage(nk, buf ^ 1);          // overlap with WMMAs below
        }

        FragU af[2];
#pragma unroll
        for (int mi = 0; mi < 2; ++mi) {
            const uint4* p = reinterpret_cast<const uint4*>(
                &As[buf][mq * 32 + mi * 16 + lr][hl * 8]);
            af[mi].q[0] = p[0];          // K = hl*8 .. +7
            af[mi].q[1] = p[2];          // K = 16+hl*8 .. +7
        }
        FragU bg[2], bu[2];
#pragma unroll
        for (int ni = 0; ni < 2; ++ni) {
            const uint4* pg = reinterpret_cast<const uint4*>(
                &Bs[buf][nq * 32 + ni * 16 + lr][hl * 8]);
            bg[ni].q[0] = pg[0]; bg[ni].q[1] = pg[2];
            const uint4* pu = reinterpret_cast<const uint4*>(
                &Bs[buf][64 + nq * 32 + ni * 16 + lr][hl * 8]);
            bu[ni].q[0] = pu[0]; bu[ni].q[1] = pu[2];
        }
#pragma unroll
        for (int mi = 0; mi < 2; ++mi)
#pragma unroll
            for (int ni = 0; ni < 2; ++ni) {
                accg[mi][ni] = __builtin_amdgcn_wmma_f32_16x16x32_bf16(
                    false, af[mi].v, false, bg[ni].v, (short)0, accg[mi][ni],
                    false, false);
                accu[mi][ni] = __builtin_amdgcn_wmma_f32_16x16x32_bf16(
                    false, af[mi].v, false, bu[ni].v, (short)0, accu[mi][ni],
                    false, false);
            }
        buf ^= 1;
    }

    // fused epilogue: bias + clamp + SiLU gating -> bf16 act
#pragma unroll
    for (int mi = 0; mi < 2; ++mi) {
#pragma unroll
        for (int ni = 0; ni < 2; ++ni) {
            const int gcol = n0 + nq * 32 + ni * 16 + lr;
            const float bg_ = bgu[(size_t)e * 2 * I_DIM + gcol];
            const float bu_ = bgu[(size_t)e * 2 * I_DIM + I_DIM + gcol];
#pragma unroll
            for (int i = 0; i < 8; ++i) {
                const int rowL = mq * 32 + mi * 16 + hl * 8 + i;
                if (rowL < rows_valid) {
                    float g = accg[mi][ni][i] + bg_;
                    float u = accu[mi][ni][i] + bu_;
                    g = fminf(g, LIMIT);
                    u = fminf(fmaxf(u, -LIMIT), LIMIT);
                    float glu = g / (1.0f + __expf(-ALPHA * g));
                    float a = (u + 1.0f) * glu;
                    act[(size_t)(mbase + rowL) * I_DIM + gcol] =
                        (unsigned short)bfround(a);
                }
            }
        }
    }
}

// ---------------------------------------------------------------------------
// Kernel 4: down GEMM + weighted scatter-add into output (f32 atomics).
// A operand is already bf16 -> staged via GLOBAL_LOAD_ASYNC_TO_LDS_B128
// (ASYNCcnt) when available. Double-buffered LDS, one barrier per K-step.
// ---------------------------------------------------------------------------
__global__ void __launch_bounds__(256)
moe_gemm_down_kernel(const unsigned short* __restrict__ act,
                     const float* __restrict__ wd,
                     const float* __restrict__ bd,
                     const int* __restrict__ offs,
                     const int* __restrict__ slot_token,
                     const float* __restrict__ slot_w,
                     float* __restrict__ out) {
    const int e  = blockIdx.z;
    const int mt = blockIdx.y;
    const int n0 = blockIdx.x * 64;

    const int base = offs[e];
    const int ce   = offs[e + 1] - base;
    if (mt * 128 >= ce) return;
    const int mbase = base + mt * 128;
    const int rows_valid = min(128, ce - mt * 128);

    __shared__ unsigned short As[2][128][40];
    __shared__ unsigned short Bs[2][64][40];

    const int ltid  = threadIdx.x;
    const int lrow  = ltid >> 1;
    const int lhalf = ltid & 1;
    const int brow  = ltid >> 2;
    const int bq    = ltid & 3;

    const int arow_clamped = min(mbase + lrow, T_TOK * 2 - 1);
    const unsigned short* aSrc =
        act + (size_t)arow_clamped * I_DIM + lhalf * 16;
    const float* bSrc = wd + ((size_t)e * H_DIM + n0 + brow) * I_DIM + bq * 8;

    const int wv   = ltid >> 5;
    const int lane = ltid & 31;
    const int mq   = wv & 3;
    const int nq   = wv >> 2;
    const int lr   = lane & 15;
    const int hl   = lane >> 4;

    auto stage = [&](int k0, int b) {
#ifdef USE_ASYNC_LDS
        // bf16 act: direct async global->LDS, bypasses VGPRs, ASYNCcnt-tracked
        async_cp16(aSrc + k0, &As[b][lrow][lhalf * 16]);
        async_cp16(aSrc + k0 + 8, &As[b][lrow][lhalf * 16 + 8]);
#else
        {
            const uint4* s = reinterpret_cast<const uint4*>(aSrc + k0);
            uint4 q0 = s[0], q1 = s[1];
            uint4* d = reinterpret_cast<uint4*>(&As[b][lrow][lhalf * 16]);
            d[0] = q0; d[1] = q1;
        }
#endif
        {
            const float4* s = reinterpret_cast<const float4*>(bSrc + k0);
            float4 x0 = s[0], x1 = s[1];
            uint4 p = make_uint4(pk2(x0.x, x0.y), pk2(x0.z, x0.w),
                                 pk2(x1.x, x1.y), pk2(x1.z, x1.w));
            *reinterpret_cast<uint4*>(&Bs[b][brow][bq * 8]) = p;
        }
    };

    v8f acc[2][2] = {};

    stage(0, 0);
    int buf = 0;
    for (int k0 = 0; k0 < I_DIM; k0 += 32) {
#ifdef USE_ASYNC_LDS
        async_wait0();                   // this wave's async LDS writes done
#endif
        __syncthreads();                 // cross-wave: tile 'buf' ready
        const int nk = k0 + 32;
        if (nk < I_DIM) {
            __builtin_prefetch(bSrc + nk + 32, 0, 1);
            stage(nk, buf ^ 1);
        }

        FragU af[2], bf[2];
#pragma unroll
        for (int mi = 0; mi < 2; ++mi) {
            const uint4* p = reinterpret_cast<const uint4*>(
                &As[buf][mq * 32 + mi * 16 + lr][hl * 8]);
            af[mi].q[0] = p[0];
            af[mi].q[1] = p[2];
        }
#pragma unroll
        for (int ni = 0; ni < 2; ++ni) {
            const uint4* p = reinterpret_cast<const uint4*>(
                &Bs[buf][nq * 32 + ni * 16 + lr][hl * 8]);
            bf[ni].q[0] = p[0]; bf[ni].q[1] = p[2];
        }
#pragma unroll
        for (int mi = 0; mi < 2; ++mi)
#pragma unroll
            for (int ni = 0; ni < 2; ++ni)
                acc[mi][ni] = __builtin_amdgcn_wmma_f32_16x16x32_bf16(
                    false, af[mi].v, false, bf[ni].v, (short)0, acc[mi][ni],
                    false, false);
        buf ^= 1;
    }

    // epilogue: (dot + b_down) * routing_weight, scatter-add
#pragma unroll
    for (int ni = 0; ni < 2; ++ni) {
        const int hcol = n0 + nq * 32 + ni * 16 + lr;
        const float bdv = bd[(size_t)e * H_DIM + hcol];
#pragma unroll
        for (int mi = 0; mi < 2; ++mi) {
#pragma unroll
            for (int i = 0; i < 8; ++i) {
                const int rowL = mq * 32 + mi * 16 + hl * 8 + i;
                if (rowL < rows_valid) {
                    const int slot = mbase + rowL;
                    float v = (acc[mi][ni][i] + bdv) * slot_w[slot];
                    atomicAdd(out + (size_t)slot_token[slot] * H_DIM + hcol, v);
                }
            }
        }
    }
}

// ---------------------------------------------------------------------------
extern "C" void kernel_launch(void* const* d_in, const int* in_sizes, int n_in,
                              void* d_out, int out_size, void* d_ws,
                              size_t ws_size, hipStream_t stream) {
    (void)in_sizes; (void)n_in; (void)ws_size;
    const float* hs  = (const float*)d_in[0];   // [512, 2880]
    const float* rw  = (const float*)d_in[1];   // [8, 2880]
    const float* rb  = (const float*)d_in[2];   // [8]
    const float* wgu = (const float*)d_in[3];   // [8, 5760, 2880]
    const float* bgu = (const float*)d_in[4];   // [8, 5760]
    const float* wd  = (const float*)d_in[5];   // [8, 2880, 2880]
    const float* bd  = (const float*)d_in[6];   // [8, 2880]
    float* out = (float*)d_out;                 // [512, 2880]

    char* ws = (char*)d_ws;
    int*            topk_idx   = (int*)(ws + 0);
    float*          topk_w     = (float*)(ws + 4096);
    int*            offs       = (int*)(ws + 8192);
    int*            slot_token = (int*)(ws + 12288);
    float*          slot_w     = (float*)(ws + 16384);
    unsigned short* act        = (unsigned short*)(ws + 32768);

    (void)hipMemsetAsync(out, 0, (size_t)out_size * sizeof(float), stream);

    moe_router_kernel<<<T_TOK / 8, 256, 0, stream>>>(hs, rw, rb,
                                                     topk_idx, topk_w);
    moe_build_lists_kernel<<<1, 256, 0, stream>>>(topk_idx, topk_w, offs,
                                                  slot_token, slot_w);
    moe_gemm_gateup_kernel<<<dim3(I_DIM / 64, 4, E_NUM), 256, 0, stream>>>(
        hs, wgu, bgu, offs, slot_token, act);
    moe_gemm_down_kernel<<<dim3(H_DIM / 64, 4, E_NUM), 256, 0, stream>>>(
        act, wd, bd, offs, slot_token, slot_w, out);
}
